// CRN_23373212025358
// MI455X (gfx1250) — compile-verified, matrix operand order checked by reading, not run on hardware
//
#include <hip/hip_runtime.h>
#include <math.h>

// ---------------------------------------------------------------------------
// CRU recurrent network for MI455X (gfx1250, wave32, WMMA bf16 16x16x32).
// T=256, B=128, I=H=512.  K = 2I = 2H = 1024.  NC = 2048 gate cols + 512 mod.
// ---------------------------------------------------------------------------

typedef __bf16 bf16_t;
typedef __bf16 bf16x16 __attribute__((ext_vector_type(16)));
typedef float  f32x8   __attribute__((ext_vector_type(8)));
typedef unsigned int u32x4 __attribute__((ext_vector_type(4)));
typedef unsigned int su32x4v __attribute__((ext_vector_type(4)));
typedef unsigned int su32x8v __attribute__((ext_vector_type(8)));

#define T_STEPS 256
#define B_ROWS  128
#define KDIM    1024      // 2*I == 2*H
#define GG      2048      // 2*G (gate+cand, real|imag)
#define NC      2560      // GG + H (mod head)
#define HDIM    512
#define H2      1024      // 2*H (state width)
#define MROWS   32768     // T*B
#define NWG_REC 80        // persistent workgroups: 2560/32 column tiles

union FragU {
    bf16x16 v;
    u32x4   u[2];
};

// Load a 16x32 bf16 fragment (A or B^T) from row-major memory, ld in elements.
// Layout per CDNA5 ISA 7.12.2: lanes 0-15 = rows, lanes 16-31 carry the K+8
// half; each half is 8 consecutive K values -> two contiguous 16B loads.
__device__ __forceinline__ bf16x16 load_frag_rm(const bf16_t* __restrict__ base, int ld) {
    const int lane = threadIdx.x & 31;
    const int m    = lane & 15;
    const int h8   = (lane >> 4) * 8;
    FragU f;
    f.u[0] = *reinterpret_cast<const u32x4*>(base + (size_t)m * ld + h8);
    f.u[1] = *reinterpret_cast<const u32x4*>(base + (size_t)m * ld + 16 + h8);
    return f.v;
}

__device__ __forceinline__ f32x8 wmma_bf16(bf16x16 a, bf16x16 b, f32x8 c) {
    return __builtin_amdgcn_wmma_f32_16x16x32_bf16(
        /*neg_a=*/false, a, /*neg_b=*/false, b,
        /*c_mod=*/(short)0, c, /*reuse_a=*/false, /*reuse_b=*/false);
}

// 16B global -> LDS async copy (ASYNCcnt tracked, no VGPR round trip).
__device__ __forceinline__ void async_to_lds_b128(void* lds, const void* g) {
    const unsigned l = (unsigned)(uintptr_t)lds;                 // LDS byte address
    const unsigned long long ga = (unsigned long long)(uintptr_t)g;
    asm volatile("global_load_async_to_lds_b128 %0, %1, off" :: "v"(l), "v"(ga) : "memory");
}
__device__ __forceinline__ void wait_async0() {
    asm volatile("s_wait_asynccnt 0x0" ::: "memory");
}

__device__ __forceinline__ void grid_sync(unsigned* cnt, unsigned target) {
    __syncthreads();
    if (threadIdx.x == 0) {
        __threadfence();
        atomicAdd(cnt, 1u);
        while (__hip_atomic_load(cnt, __ATOMIC_RELAXED, __HIP_MEMORY_SCOPE_AGENT) < target) {
            __builtin_amdgcn_s_sleep(2);
        }
    }
    __syncthreads();
    __threadfence();
}

// ---------------------------------------------------------------------------
// Weight fusion (transposed / N-major, bf16): Wt[c][r] from the complex block
// [[Wr, Wi], [-Wi, Wr]] plus the mod head.
// ---------------------------------------------------------------------------
__global__ void build_w_kernel(const float* __restrict__ wihr, const float* __restrict__ wihi,
                               const float* __restrict__ whhr, const float* __restrict__ whhi,
                               const float* __restrict__ wihm, const float* __restrict__ whhm,
                               bf16_t* __restrict__ WtIH, bf16_t* __restrict__ WtHH) {
    const long idx = (long)blockIdx.x * blockDim.x + threadIdx.x;
    if (idx >= (long)NC * KDIM) return;
    const int c = (int)(idx / KDIM);
    const int r = (int)(idx % KDIM);
    float vih, vhh;
    if (c < GG) {
        const int rr = r & (HDIM - 1);
        const int cc = (c < 1024) ? c : c - 1024;
        const bool rt = (r < HDIM);
        const bool cl = (c < 1024);
        const long wi = (long)rr * 1024 + cc;
        if (rt && cl)        { vih =  wihr[wi]; vhh =  whhr[wi]; }
        else if (!rt && cl)  { vih = -wihi[wi]; vhh = -whhi[wi]; }
        else if (rt)         { vih =  wihi[wi]; vhh =  whhi[wi]; }
        else                 { vih =  wihr[wi]; vhh =  whhr[wi]; }
    } else {
        const int cc = c - GG;
        vih = wihm[(long)r * HDIM + cc];
        vhh = whhm[(long)r * HDIM + cc];
    }
    WtIH[idx] = (bf16_t)vih;
    WtHH[idx] = (bf16_t)vhh;
}

__global__ void build_bias_kernel(const float* __restrict__ bih, const float* __restrict__ bhh,
                                  const float* __restrict__ bihm, const float* __restrict__ bhhm,
                                  float* __restrict__ bias) {
    const int c = blockIdx.x * blockDim.x + threadIdx.x;
    if (c >= NC) return;
    bias[c] = (c < GG) ? (bih[c] + bhh[c]) : (bihm[c - GG] + bhhm[c - GG]);
}

__global__ void convert_x_kernel(const float* __restrict__ x, bf16_t* __restrict__ xb) {
    const long i = ((long)blockIdx.x * blockDim.x + threadIdx.x) * 4;
    if (i + 3 < (long)MROWS * KDIM) {
        xb[i + 0] = (bf16_t)x[i + 0];
        xb[i + 1] = (bf16_t)x[i + 1];
        xb[i + 2] = (bf16_t)x[i + 2];
        xb[i + 3] = (bf16_t)x[i + 3];
    }
}

__global__ void init_h_kernel(const float* __restrict__ hx, float* __restrict__ h,
                              bf16_t* __restrict__ hb) {
    const int i = blockIdx.x * blockDim.x + threadIdx.x;
    if (i < B_ROWS * H2) {
        const float v = hx[i];
        h[i]  = v;
        hb[i] = (bf16_t)v;
    }
}

// ---------------------------------------------------------------------------
// Hoisted input GEMM: pre_x(32768 x 2560) = Xb(32768 x 1024) @ WtIH^T + bias.
// 128x128 block tile, 4 waves of 64x64; K chunks of 32 staged into LDS with
// GLOBAL_LOAD_ASYNC_TO_LDS_B128 (ASYNCcnt), consumed by 16 bf16 WMMAs.
// ---------------------------------------------------------------------------
__global__ void __launch_bounds__(128)
gemm_prex_kernel(const bf16_t* __restrict__ X, const bf16_t* __restrict__ Wt,
                 const float* __restrict__ bias, float* __restrict__ pre) {
    __shared__ bf16_t As[128 * 32];
    __shared__ bf16_t Bs[128 * 32];
    const int m0 = blockIdx.x * 128;
    const int n0 = blockIdx.y * 128;
    const int t  = threadIdx.x;
    const int wave = t >> 5;
    const int wr = wave >> 1;
    const int wc = wave & 1;
    const int lane = t & 31;
    const int ln = lane & 15;
    const int mo = (lane >> 4) * 8;

    f32x8 acc[4][4];
#pragma unroll
    for (int i = 0; i < 4; ++i)
#pragma unroll
        for (int j = 0; j < 4; ++j)
            acc[i][j] = (f32x8){0.f, 0.f, 0.f, 0.f, 0.f, 0.f, 0.f, 0.f};

    const bf16_t* arow = X  + (size_t)(m0 + t) * KDIM;
    const bf16_t* brow = Wt + (size_t)(n0 + t) * KDIM;

    for (int k0 = 0; k0 < KDIM; k0 += 32) {
        __syncthreads();
#pragma unroll
        for (int i = 0; i < 4; ++i) {
            async_to_lds_b128(As + t * 32 + i * 8, arow + k0 + i * 8);
            async_to_lds_b128(Bs + t * 32 + i * 8, brow + k0 + i * 8);
        }
        if (k0 + 32 < KDIM) {
            __builtin_prefetch(arow + k0 + 32, 0, 1);
            __builtin_prefetch(brow + k0 + 32, 0, 1);
        }
        wait_async0();
        __syncthreads();

        bf16x16 a[4], b[4];
#pragma unroll
        for (int i = 0; i < 4; ++i) a[i] = load_frag_rm(As + (wr * 64 + 16 * i) * 32, 32);
#pragma unroll
        for (int j = 0; j < 4; ++j) b[j] = load_frag_rm(Bs + (wc * 64 + 16 * j) * 32, 32);
#pragma unroll
        for (int i = 0; i < 4; ++i)
#pragma unroll
            for (int j = 0; j < 4; ++j)
                acc[i][j] = wmma_bf16(a[i], b[j], acc[i][j]);
    }

#pragma unroll
    for (int i = 0; i < 4; ++i) {
#pragma unroll
        for (int j = 0; j < 4; ++j) {
            const int row0 = m0 + wr * 64 + 16 * i;
            const int col0 = n0 + wc * 64 + 16 * j;
            const float bb = bias[col0 + ln];
            float* dst = pre + (size_t)row0 * NC + col0;
#pragma unroll
            for (int q = 0; q < 8; ++q)
                dst[(size_t)(mo + q) * NC + ln] = acc[i][j][q] + bb;
        }
    }
}

// ---------------------------------------------------------------------------
// Persistent recurrence, 80 WGs x 128 threads.
//  - Weight slice (32 cols x 1024 K, 64KB bf16) preloaded to LDS once.
//  - Per step, wave 0 issues a TDM tensor_load_to_lds of the strided pre_x
//    2D tile (128 x 32 fp32, stride 2560) which overlaps the WMMA K-loop.
//  Phase 1: P = pre_x[t] + h_bf16 @ W_hh   (WMMA, B-frags from LDS)
//  Phase 2: WGs 0..63, 2 rows each: softmax(mod) + sigmoid/tanh gating.
// ---------------------------------------------------------------------------
__global__ void __launch_bounds__(128)
recurrent_kernel(const float* __restrict__ pre, const bf16_t* __restrict__ WtHH,
                 float* __restrict__ h, bf16_t* __restrict__ hb,
                 float* __restrict__ P, float* __restrict__ out,
                 unsigned* __restrict__ cnt) {
    __shared__ bf16_t Bsh[32 * KDIM];     // 64 KB: this WG's weight columns
    __shared__ float  PreT[B_ROWS * 32];  // 16 KB: TDM-staged pre_x tile
    __shared__ float  red[128];
    const int wg   = blockIdx.x;
    const int t    = threadIdx.x;
    const int wave = t >> 5;
    const int lane = t & 31;
    const int ln = lane & 15;
    const int mo = (lane >> 4) * 8;
    const int n0 = wg * 32;       // column tile of 32 in [0, 2560)
    const int m0 = wave * 32;     // row tile of 32 in [0, 128)
    unsigned bar = 0;

    // ---- One-time weight preload into LDS (async-to-LDS, 512B/thread) ----
    {
        const int row  = t >> 2;  // 0..31
        const int part = t & 3;   // 0..3, 256 elements each
        const bf16_t* src = WtHH + (size_t)(n0 + row) * KDIM + part * 256;
        bf16_t* dst = Bsh + (size_t)row * KDIM + part * 256;
#pragma unroll
        for (int i = 0; i < 32; ++i)
            async_to_lds_b128(dst + i * 8, src + i * 8);
        wait_async0();
        __syncthreads();
    }

    for (int step = 0; step < T_STEPS; ++step) {
        // ---- Issue TDM load of pre_x tile: 128 rows x 32 f32, stride 2560 ----
        if (wave == 0) {
            const unsigned long long ga =
                (unsigned long long)(uintptr_t)(pre + (size_t)step * B_ROWS * NC + n0);
            su32x4v g0;
            g0[0] = 1u;                                   // count=1, user descriptor
            g0[1] = (unsigned)(uintptr_t)PreT;            // lds_addr
            g0[2] = (unsigned)ga;                         // global_addr[31:0]
            g0[3] = ((unsigned)(ga >> 32) & 0x01FFFFFFu) | (2u << 30); // addr hi + type=2
            su32x8v g1;
            g1[0] = (2u << 16);                           // data_size = 4B
            g1[1] = ((unsigned)NC << 16);                 // tensor_dim0 = 2560 (lo16)
            g1[2] = ((unsigned)B_ROWS << 16);             // dim0 hi=0 | tensor_dim1 = 128
            g1[3] = (32u << 16);                          // dim1 hi=0 | tile_dim0 = 32
            g1[4] = (unsigned)B_ROWS;                     // tile_dim1 = 128, tile_dim2 = 0
            g1[5] = (unsigned)NC;                         // tensor_dim0_stride = 2560
            g1[6] = 0u;
            g1[7] = 0u;
            asm volatile("tensor_load_to_lds %0, %1" :: "s"(g0), "s"(g1) : "memory");
        }

        // ---- Phase 1: recurrent GEMM tile (32x32 per wave, 2x2 WMMA) ----
        f32x8 acc[2][2];
#pragma unroll
        for (int i = 0; i < 2; ++i)
#pragma unroll
            for (int j = 0; j < 2; ++j)
                acc[i][j] = (f32x8){0.f, 0.f, 0.f, 0.f, 0.f, 0.f, 0.f, 0.f};

        for (int k0 = 0; k0 < KDIM; k0 += 32) {
            bf16x16 a0 = load_frag_rm(hb + (size_t)(m0)      * KDIM + k0, KDIM);
            bf16x16 a1 = load_frag_rm(hb + (size_t)(m0 + 16) * KDIM + k0, KDIM);
            bf16x16 b0 = load_frag_rm(Bsh + k0, KDIM);
            bf16x16 b1 = load_frag_rm(Bsh + (size_t)16 * KDIM + k0, KDIM);
            acc[0][0] = wmma_bf16(a0, b0, acc[0][0]);
            acc[0][1] = wmma_bf16(a0, b1, acc[0][1]);
            acc[1][0] = wmma_bf16(a1, b0, acc[1][0]);
            acc[1][1] = wmma_bf16(a1, b1, acc[1][1]);
        }

        // TDM tile must be complete before the epilogue consumes it.
        if (wave == 0) __builtin_amdgcn_s_wait_tensorcnt(0);
        __syncthreads();

#pragma unroll
        for (int i = 0; i < 2; ++i)
#pragma unroll
            for (int j = 0; j < 2; ++j) {
                const int r0 = m0 + 16 * i + mo;
                const int cl = 16 * j + ln;       // local column 0..31
#pragma unroll
                for (int q = 0; q < 8; ++q) {
                    const int r = r0 + q;
                    P[(size_t)r * NC + n0 + cl] = acc[i][j][q] + PreT[r * 32 + cl];
                }
            }
        ++bar;
        grid_sync(cnt, bar * NWG_REC);

        // ---- Phase 2: softmax + gating, 2 batch rows per WG (WGs 0..63) ----
        if (wg < 64) {
            const int r   = wg * 2 + (t >> 6);
            const int tid = t & 63;
            const int rb  = (t >> 6) * 64;
            const float* Pr = P + (size_t)r * NC;

            float mreg[8];
            float lmax = -3.4e38f;
#pragma unroll
            for (int i = 0; i < 8; ++i) {
                mreg[i] = Pr[GG + tid + 64 * i];
                lmax = fmaxf(lmax, mreg[i]);
            }
            red[t] = lmax;
            __syncthreads();
            float rmax = -3.4e38f;
            for (int k = 0; k < 64; ++k) rmax = fmaxf(rmax, red[rb + k]);
            float lsum = 0.f;
#pragma unroll
            for (int i = 0; i < 8; ++i) {
                mreg[i] = __expf(mreg[i] - rmax);
                lsum += mreg[i];
            }
            __syncthreads();
            red[t] = lsum;
            __syncthreads();
            float rsum = 0.f;
            for (int k = 0; k < 64; ++k) rsum += red[rb + k];
            const float inv = 1.0f / rsum;
#pragma unroll
            for (int i = 0; i < 8; ++i) mreg[i] *= inv;

            float*  hr   = h   + (size_t)r * H2;
            bf16_t* hbr  = hb  + (size_t)r * H2;
            float*  outr = out + ((size_t)step * B_ROWS + r) * H2;
#pragma unroll
            for (int i = 0; i < 16; ++i) {
                const int j  = tid + 64 * i;
                const int jj = (j < HDIM) ? j : (j - HDIM);
                const float gpre = Pr[(j < HDIM) ? j          : (1024 + jj)];
                const float cpre = Pr[(j < HDIM) ? (HDIM + j) : (1536 + jj)];
                const float g = 1.0f / (1.0f + __expf(-gpre));
                const float c = tanhf(cpre);
                const float hn = mreg[i & 7] * (g * hr[j] + (1.0f - g) * c);
                hr[j]  = hn;
                hbr[j] = (bf16_t)hn;
                outr[j] = hn;
                if (step == T_STEPS - 1)
                    out[(size_t)T_STEPS * B_ROWS * H2 + (size_t)r * H2 + j] = hn;
            }
        }
        ++bar;
        grid_sync(cnt, bar * NWG_REC);
    }
}

// ---------------------------------------------------------------------------
// Host-side launch
// ---------------------------------------------------------------------------
static inline void* ws_alloc(char* base, size_t& off, size_t bytes) {
    void* p = base + off;
    off += (bytes + 255) & ~(size_t)255;
    return p;
}

extern "C" void kernel_launch(void* const* d_in, const int* in_sizes, int n_in,
                              void* d_out, int out_size, void* d_ws, size_t ws_size,
                              hipStream_t stream) {
    const float* x     = (const float*)d_in[0];
    const float* hx    = (const float*)d_in[1];
    const float* wihr  = (const float*)d_in[2];
    const float* wihi  = (const float*)d_in[3];
    const float* whhr  = (const float*)d_in[4];
    const float* whhi  = (const float*)d_in[5];
    const float* bih   = (const float*)d_in[6];
    const float* bhh   = (const float*)d_in[7];
    const float* wihm  = (const float*)d_in[8];
    const float* whhm  = (const float*)d_in[9];
    const float* bihm  = (const float*)d_in[10];
    const float* bhhm  = (const float*)d_in[11];
    float* out = (float*)d_out;

    char* base = (char*)d_ws;
    size_t off = 0;
    unsigned* cnt  = (unsigned*)ws_alloc(base, off, 256);
    bf16_t* WtIH   = (bf16_t*)ws_alloc(base, off, (size_t)NC * KDIM * sizeof(bf16_t));
    bf16_t* WtHH   = (bf16_t*)ws_alloc(base, off, (size_t)NC * KDIM * sizeof(bf16_t));
    float*  bias   = (float*) ws_alloc(base, off, (size_t)NC * sizeof(float));
    bf16_t* Xb     = (bf16_t*)ws_alloc(base, off, (size_t)MROWS * KDIM * sizeof(bf16_t));
    float*  preX   = (float*) ws_alloc(base, off, (size_t)MROWS * NC * sizeof(float));
    float*  h      = (float*) ws_alloc(base, off, (size_t)B_ROWS * H2 * sizeof(float));
    bf16_t* hb     = (bf16_t*)ws_alloc(base, off, (size_t)B_ROWS * H2 * sizeof(bf16_t));
    float*  P      = (float*) ws_alloc(base, off, (size_t)B_ROWS * NC * sizeof(float));
    (void)ws_size; (void)in_sizes; (void)n_in; (void)out_size;

    hipMemsetAsync(cnt, 0, 256, stream);

    {
        const long n = (long)NC * KDIM;
        build_w_kernel<<<(unsigned)((n + 255) / 256), 256, 0, stream>>>(
            wihr, wihi, whhr, whhi, wihm, whhm, WtIH, WtHH);
    }
    build_bias_kernel<<<(NC + 255) / 256, 256, 0, stream>>>(bih, bhh, bihm, bhhm, bias);
    {
        const long n = (long)MROWS * KDIM;       // 33.5M elems, 4 per thread
        convert_x_kernel<<<(unsigned)(n / 4 / 256), 256, 0, stream>>>(x, Xb);
    }
    gemm_prex_kernel<<<dim3(MROWS / 128, NC / 128), 128, 0, stream>>>(Xb, WtIH, bias, preX);
    init_h_kernel<<<(B_ROWS * H2 + 255) / 256, 256, 0, stream>>>(hx, h, hb);
    recurrent_kernel<<<NWG_REC, 128, 0, stream>>>(preX, WtHH, h, hb, P, out, cnt);
}
